// cere_SNN_64768106824237
// MI455X (gfx1250) — matrix operand order repaired and checked by reading, not compiled, not run on hardware
//
#include <hip/hip_runtime.h>

typedef __bf16 bf16;
typedef __attribute__((ext_vector_type(16))) __bf16 v16bf;
typedef __attribute__((ext_vector_type(8)))  __bf16 v8bf;
typedef __attribute__((ext_vector_type(8)))  float  v8f;
typedef __attribute__((ext_vector_type(4)))  int    i32x4;

#define DECAY   0.2f
#define THRESH  0.5f
#define BSZ     512
#define IN_MF   1024
#define NMF     1024
#define NGC     8192
#define NPC     1024
#define NDCN    256
#define TSTEPS  50

#define BM 64
#define BN 64
#define BK 64
#define RS (BK + 8)      // padded LDS row stride (elements): 144B -> conflict-friendly

#ifndef __has_builtin
#define __has_builtin(x) 0
#endif
#if __has_builtin(__builtin_amdgcn_global_load_async_to_lds_b128)
#define HAVE_ASYNC 1
#else
#define HAVE_ASYNC 0
#endif

// ---------------------------------------------------------------------------
// 16B global->LDS copy: async (ASYNCcnt) on gfx1250, sync fallback otherwise
// ---------------------------------------------------------------------------
static __device__ __forceinline__ void cp16(const bf16* __restrict__ g, bf16* l) {
#if HAVE_ASYNC
    __builtin_amdgcn_global_load_async_to_lds_b128(
        (__attribute__((address_space(1))) i32x4*)(bf16*)g,
        (__attribute__((address_space(3))) i32x4*)l,
        0, 0);
#else
    *(v8bf*)l = *(const v8bf*)g;
#endif
}

static __device__ __forceinline__ void wait_async_done() {
#if HAVE_ASYNC
#if __has_builtin(__builtin_amdgcn_s_wait_asynccnt)
    __builtin_amdgcn_s_wait_asynccnt(0);
#else
    asm volatile("s_wait_asynccnt 0" ::: "memory");
#endif
#endif
}

// ---------------------------------------------------------------------------
// Fragment loader per CDNA5 WMMA VGPR layouts (ISA 7.12.2); works on LDS ptrs
// ---------------------------------------------------------------------------
static __device__ __forceinline__ v16bf load_frag(const bf16* p, int hi_off) {
    v8bf lo = *(const v8bf*)p;
    v8bf hi = *(const v8bf*)(p + hi_off);
    v16bf r;
#pragma unroll
    for (int i = 0; i < 8; ++i) { r[i] = lo[i]; r[i + 8] = hi[i]; }
    return r;
}

static __device__ __forceinline__ v8f wmma_bf16(v16bf a, v16bf b, v8f c) {
    return __builtin_amdgcn_wmma_f32_16x16x32_bf16(false, a, false, b, (short)0, c, false, false);
}

// ---------------------------------------------------------------------------
// Fused LIF GEMM:  Hm_new = (Hm>TH ? 0 : DECAY*Hm) + A @ W^T + bias
//   LAYER==1: Sout=spike(bf16), wave-reduced spike count atomicAdd -> cnt
//   LAYER==2: Sout=spike(bf16)
//   LAYER==3: Sout = bf16(|cntIn|/(B*NMF) - spike)            (din)
//   LAYER==4: acc[idx] += spike                               (d_out)
// A: [512,K] bf16 row-major, W: [N,K] bf16 row-major. M=512 fixed.
// Block 64x64, 128 threads (4 waves 2x2), wave tile 32x32 (2x2 WMMA frags).
// Double-buffered async global->LDS staging over BK=64 chunks.
// ---------------------------------------------------------------------------
template <int LAYER>
__global__ __launch_bounds__(128) void snn_gemm(
    const bf16* __restrict__ A, const bf16* __restrict__ W,
    const float* __restrict__ bias, float* __restrict__ Hm,
    bf16* __restrict__ Sout, float* __restrict__ cnt,
    const float* __restrict__ cntIn, float* __restrict__ acc,
    int N, int K)
{
    __shared__ __align__(16) bf16 sA[2][BM][RS];
    __shared__ __align__(16) bf16 sB[2][BN][RS];

    const int tid  = threadIdx.x;
    const int lane = tid & 31;
    const int wave = tid >> 5;           // 0..3
    const int wm = wave >> 1;            // 0..1 along M
    const int wn = wave & 1;             // 0..1 along N
    const int bM = blockIdx.x * BM;
    const int bN = blockIdx.y * BN;
    const int m0 = bM + wm * 32;
    const int n0 = bN + wn * 32;
    const int l15 = lane & 15;
    const int ahalf = (lane & 16) ? 8  : 0;   // A: per-lane K chunk offset
    const int bhalf = (lane & 16) ? 8  : 0;   // B: per-lane K chunk offset

    // Stage one BK-wide tile of A (64xBK) and W (64xBK) into LDS buffer `buf`.
    auto stage = [&](int buf, int k0) {
#pragma unroll
        for (int rep = 0; rep < 4; ++rep) {
            const int cid = tid + rep * 128;          // 512 chunks of 16B
            const int r = cid >> 3;
            const int c = (cid & 7) * 8;
            cp16(A + (size_t)(bM + r) * K + k0 + c, &sA[buf][r][c]);
            cp16(W + (size_t)(bN + r) * K + k0 + c, &sB[buf][r][c]);
        }
    };

    v8f c00 = {}, c01 = {}, c10 = {}, c11 = {};

    stage(0, 0);
    int cur = 0;
    for (int k = 0; k < K; k += BK) {
        wait_async_done();
        __syncthreads();                 // tile `cur` visible; prev reads done
        if (k + BK < K) stage(cur ^ 1, k + BK);

        const bf16* la0 = &sA[cur][wm * 32 + l15][ahalf];
        const bf16* la1 = la0 + 16 * RS;
        const bf16* lb0 = &sB[cur][wn * 32 + l15][bhalf];
        const bf16* lb1 = lb0 + 16 * RS;
#pragma unroll
        for (int kk = 0; kk < BK; kk += 32) {
            v16bf a0 = load_frag(la0 + kk, 16);
            v16bf a1 = load_frag(la1 + kk, 16);
            v16bf b0 = load_frag(lb0 + kk, 16);
            v16bf b1 = load_frag(lb1 + kk, 16);
            c00 = wmma_bf16(a0, b0, c00);
            c01 = wmma_bf16(a0, b1, c01);
            c10 = wmma_bf16(a1, b0, c10);
            c11 = wmma_bf16(a1, b1, c11);
        }
        cur ^= 1;
    }

    v8f cc[2][2] = {{c00, c01}, {c10, c11}};

    int spikes = 0;
    float meanabs = 0.0f;
    if (LAYER == 3)
        meanabs = fabsf(cntIn[0]) * (1.0f / ((float)BSZ * (float)NMF));

#pragma unroll
    for (int i = 0; i < 2; ++i) {
        const int rowb = m0 + i * 16 + ((lane & 16) ? 8 : 0);
#pragma unroll
        for (int j = 0; j < 2; ++j) {
            const int col = n0 + j * 16 + l15;
            const float bia = bias[col];
#pragma unroll
            for (int v = 0; v < 8; ++v) {
                const size_t idx = (size_t)(rowb + v) * N + col;
                const float old = Hm[idx];
                const float nm  = (old > THRESH ? 0.0f : DECAY * old) + cc[i][j][v] + bia;
                Hm[idx] = nm;
                const float s = nm > THRESH ? 1.0f : 0.0f;
                if (LAYER == 1)      { Sout[idx] = (bf16)s; spikes += (nm > THRESH); }
                else if (LAYER == 2) { Sout[idx] = (bf16)s; }
                else if (LAYER == 3) { Sout[idx] = (bf16)(meanabs - s); }
                else                 { acc[idx] += s; }
            }
        }
    }

    if (LAYER == 1) {
#pragma unroll
        for (int off = 16; off; off >>= 1) spikes += __shfl_xor(spikes, off, 32);
        if (lane == 0) atomicAdd(cnt, (float)spikes);
    }
}

// ---------------------------------------------------------------------------
// Prep kernels
// ---------------------------------------------------------------------------
__global__ void zero_f32(float* __restrict__ p, int n) {
    int i = blockIdx.x * blockDim.x + threadIdx.x;
    if (i < n) p[i] = 0.0f;
}

__global__ void cvt_w(const float* __restrict__ w, bf16* __restrict__ o, int n) {
    int i = blockIdx.x * blockDim.x + threadIdx.x;
    if (i < n) o[i] = (bf16)w[i];
}

__global__ void cvt_w_mask(const float* __restrict__ w, const float* __restrict__ msk,
                           bf16* __restrict__ o, int n) {
    int i = blockIdx.x * blockDim.x + threadIdx.x;
    if (i < n) o[i] = (bf16)(w[i] * msk[i]);
}

// xs[t][b][i] (bf16) from m/u/sigma/f each [B,256,T] fp32
__global__ void pack_x(const float* __restrict__ m, const float* __restrict__ u,
                       const float* __restrict__ s, const float* __restrict__ f,
                       bf16* __restrict__ xs) {
    int idx = blockIdx.x * blockDim.x + threadIdx.x;
    if (idx >= TSTEPS * BSZ * IN_MF) return;
    const int i = idx & 1023;
    const int b = (idx >> 10) & 511;
    const int t = idx >> 19;                       // 512*1024 = 2^19
    const float* src = (i < 256) ? m : (i < 512) ? u : (i < 768) ? s : f;
    const int c = i & 255;
    xs[idx] = (bf16)src[((size_t)b * 256 + c) * TSTEPS + t];
}

// ---------------------------------------------------------------------------
extern "C" void kernel_launch(void* const* d_in, const int* in_sizes, int n_in,
                              void* d_out, int out_size, void* d_ws, size_t ws_size,
                              hipStream_t stream) {
    (void)in_sizes; (void)n_in; (void)out_size; (void)ws_size;
    const float* m     = (const float*)d_in[0];
    const float* u     = (const float*)d_in[1];
    const float* sg    = (const float*)d_in[2];
    const float* f     = (const float*)d_in[3];
    const float* fc1_w = (const float*)d_in[4];
    const float* fc1_b = (const float*)d_in[5];
    const float* fc2_w = (const float*)d_in[6];
    const float* fc2_b = (const float*)d_in[7];
    const float* fc3_w = (const float*)d_in[8];
    const float* fc3_b = (const float*)d_in[9];
    const float* fc4_w = (const float*)d_in[10];
    const float* fc4_b = (const float*)d_in[11];
    const float* mask  = (const float*)d_in[12];

    char* ws = (char*)d_ws;
    size_t off = 0;
    auto carve = [&](size_t bytes) -> char* {
        char* p = ws + off;
        off = (off + bytes + 255) & ~(size_t)255;
        return p;
    };

    bf16* W1b = (bf16*)carve((size_t)NMF  * IN_MF * 2);
    bf16* W2b = (bf16*)carve((size_t)NGC  * NMF   * 2);
    bf16* W3b = (bf16*)carve((size_t)NPC  * NGC   * 2);
    bf16* W4b = (bf16*)carve((size_t)NDCN * NPC   * 2);
    bf16* XS  = (bf16*)carve((size_t)TSTEPS * BSZ * IN_MF * 2);
    float* H1M = (float*)carve((size_t)BSZ * NMF  * 4);
    float* H2M = (float*)carve((size_t)BSZ * NGC  * 4);
    float* H3M = (float*)carve((size_t)BSZ * NPC  * 4);
    float* H4M = (float*)carve((size_t)BSZ * NDCN * 4);
    bf16*  S1  = (bf16*)carve((size_t)BSZ * NMF * 2);
    bf16*  S2  = (bf16*)carve((size_t)BSZ * NGC * 2);
    bf16*  DIN = (bf16*)carve((size_t)BSZ * NPC * 2);
    float* CNT = (float*)carve(64 * 4);

    const dim3 blk256(256);
    const dim3 blkG(128);
    auto g1 = [](int n) { return dim3((unsigned)((n + 255) / 256)); };

    // --- one-time prep (runs every call; deterministic) ---
    cvt_w     <<<g1(NMF * IN_MF),  blk256, 0, stream>>>(fc1_w, W1b, NMF * IN_MF);
    cvt_w_mask<<<g1(NGC * NMF),    blk256, 0, stream>>>(fc2_w, mask, W2b, NGC * NMF);
    cvt_w     <<<g1(NPC * NGC),    blk256, 0, stream>>>(fc3_w, W3b, NPC * NGC);
    cvt_w     <<<g1(NDCN * NPC),   blk256, 0, stream>>>(fc4_w, W4b, NDCN * NPC);
    pack_x    <<<g1(TSTEPS * BSZ * IN_MF), blk256, 0, stream>>>(m, u, sg, f, XS);

    zero_f32<<<g1(BSZ * NMF),  blk256, 0, stream>>>(H1M, BSZ * NMF);
    zero_f32<<<g1(BSZ * NGC),  blk256, 0, stream>>>(H2M, BSZ * NGC);
    zero_f32<<<g1(BSZ * NPC),  blk256, 0, stream>>>(H3M, BSZ * NPC);
    zero_f32<<<g1(BSZ * NDCN), blk256, 0, stream>>>(H4M, BSZ * NDCN);
    zero_f32<<<g1(BSZ * NDCN), blk256, 0, stream>>>((float*)d_out, BSZ * NDCN);
    zero_f32<<<1, 64, 0, stream>>>(CNT, 64);

    // --- recurrence: 4 fused WMMA GEMMs per timestep ---
    for (int t = 0; t < TSTEPS; ++t) {
        const bf16* xt = XS + (size_t)t * BSZ * IN_MF;
        snn_gemm<1><<<dim3(BSZ / BM, NMF / BN),  blkG, 0, stream>>>(
            xt,  W1b, fc1_b, H1M, S1,  CNT + t, nullptr, nullptr, NMF,  IN_MF);
        snn_gemm<2><<<dim3(BSZ / BM, NGC / BN),  blkG, 0, stream>>>(
            S1,  W2b, fc2_b, H2M, S2,  nullptr, nullptr, nullptr, NGC,  NMF);
        snn_gemm<3><<<dim3(BSZ / BM, NPC / BN),  blkG, 0, stream>>>(
            S2,  W3b, fc3_b, H3M, DIN, nullptr, CNT + t, nullptr, NPC,  NGC);
        snn_gemm<4><<<dim3(BSZ / BM, NDCN / BN), blkG, 0, stream>>>(
            DIN, W4b, fc4_b, H4M, nullptr, nullptr, nullptr, (float*)d_out, NDCN, NPC);
    }
}
